// Qwen2moeDynSparseMoeBlock_6433861010005
// MI455X (gfx1250) — matrix operand order repaired and checked by reading, not compiled, not run on hardware
//
#include <hip/hip_runtime.h>
#include <stdint.h>

#define T_TOK 4096
#define H_DIM 2048
#define E_NUM 8
#define I_DIM 1408
#define BLK   128
#define KT    32

typedef __attribute__((ext_vector_type(16))) __bf16   bf16x16;
typedef __attribute__((ext_vector_type(8)))  float    f32x8;
typedef __attribute__((ext_vector_type(4)))  uint32_t u32x4;

union FragAB { bf16x16 bf; u32x4 u[2]; };

__device__ __forceinline__ uint16_t f32_to_bf16(float f) {
    union { float f; uint32_t u; } c; c.f = f;
    uint32_t u = c.u;
    uint32_t r = u + 0x7FFFu + ((u >> 16) & 1u);   // round-to-nearest-even
    return (uint16_t)(r >> 16);
}

// ---- CDNA5 async global->LDS DMA (ASYNCcnt-tracked), 16B per lane ----
__device__ __forceinline__ void ld_async_b128(uint16_t* lds, const uint16_t* g) {
    asm volatile("global_load_async_to_lds_b128 %0, %1, off"
                 :: "v"((uint32_t)(uintptr_t)lds), "v"((uint64_t)(uintptr_t)g)
                 : "memory");
}
__device__ __forceinline__ void wait_async0() {
    asm volatile("s_wait_asynccnt 0x0" ::: "memory");
}

// ---------------- zero output ----------------
__global__ void k_zero(float* __restrict__ p, int n4) {
    int i = blockIdx.x * blockDim.x + threadIdx.x;
    int stride = gridDim.x * blockDim.x;
    float4 z = {0.f, 0.f, 0.f, 0.f};
    for (; i < n4; i += stride) ((float4*)p)[i] = z;
}

// ---------------- fp32 -> bf16 conversion ----------------
__global__ void k_cvt_bf16(const float* __restrict__ src, uint16_t* __restrict__ dst, int n4) {
    int i = blockIdx.x * blockDim.x + threadIdx.x;
    int stride = gridDim.x * blockDim.x;
    for (; i < n4; i += stride) {
        float4 v = ((const float4*)src)[i];
        ushort4 o;
        o.x = f32_to_bf16(v.x); o.y = f32_to_bf16(v.y);
        o.z = f32_to_bf16(v.z); o.w = f32_to_bf16(v.w);
        ((ushort4*)dst)[i] = o;
    }
}

// ---------------- router: logits, softmax, top-2 ----------------
// 256 threads = 8 waves, one wave per expert (wave32).
__global__ __launch_bounds__(256) void k_router(const float* __restrict__ x,
                                                const float* __restrict__ gw,
                                                float* __restrict__ logits_out,
                                                float* __restrict__ comb) {
    int t = blockIdx.x;
    int wid = threadIdx.x >> 5, lane = threadIdx.x & 31;
    __shared__ float slog[E_NUM];
    const float* xr = x + (size_t)t * H_DIM;
    const float* gr = gw + (size_t)wid * H_DIM;
    float sum = 0.f;
    for (int j = lane; j < H_DIM; j += 32) sum += xr[j] * gr[j];
    for (int off = 16; off; off >>= 1) sum += __shfl_xor(sum, off, 32);
    if (lane == 0) slog[wid] = sum;
    __syncthreads();
    if (threadIdx.x == 0) {
        float l[E_NUM], m = -1e30f;
        for (int e = 0; e < E_NUM; e++) { l[e] = slog[e]; m = fmaxf(m, l[e]); }
        float p[E_NUM], s = 0.f;
        for (int e = 0; e < E_NUM; e++) { p[e] = __expf(l[e] - m); s += p[e]; }
        for (int e = 0; e < E_NUM; e++) p[e] /= s;
        int e0 = 0;
        for (int e = 1; e < E_NUM; e++) if (p[e] > p[e0]) e0 = e;
        int e1 = (e0 == 0) ? 1 : 0;
        for (int e = 0; e < E_NUM; e++) if (e != e0 && p[e] > p[e1]) e1 = e;
        float wsum = p[e0] + p[e1];
        float w0 = p[e0] / wsum, w1 = p[e1] / wsum;
        for (int e = 0; e < E_NUM; e++) {
            logits_out[(size_t)t * E_NUM + e] = l[e];
            comb[(size_t)t * E_NUM + e] = (e == e0) ? w0 : ((e == e1) ? w1 : 0.f);
        }
    }
}

// ---------------- deterministic per-expert token compaction ----------------
__global__ void k_compact(const float* __restrict__ comb, int* __restrict__ cnt,
                          int* __restrict__ tok, float* __restrict__ wt) {
    int e = blockIdx.x, lane = threadIdx.x;
    int c = 0;
    for (int base = 0; base < T_TOK; base += 32) {
        int t = base + lane;
        float w = comb[(size_t)t * E_NUM + e];
        bool act = (w != 0.f);
        unsigned long long m = __ballot(act);
        int pre = __popcll(m & ((1ull << lane) - 1ull));
        if (act) {
            tok[e * T_TOK + c + pre] = t;
            wt[e * T_TOK + c + pre]  = w;
        }
        c += __popcll(m);
    }
    if (lane == 0) cnt[e] = c;
}

// ---------------- GEMM1: gather(x) @ [wg|wu]^T, fused SiLU*up -> act (bf16) --
// 128x128 block tile, 8 waves, each wave 2x4 wmma tiles for gate AND up.
// Double-buffered LDS fed by global_load_async_to_lds_b128.
__global__ __launch_bounds__(256) void k_gemm1(
    const uint16_t* __restrict__ xbf, const uint16_t* __restrict__ wgbf,
    const uint16_t* __restrict__ wubf, const int* __restrict__ cnt,
    const int* __restrict__ tok, uint16_t* __restrict__ act) {
  int e  = blockIdx.x % E_NUM;
  int tt = blockIdx.x / E_NUM;
  int nrows = cnt[e] - tt * BLK;
  if (nrows <= 0) return;
  if (nrows > BLK) nrows = BLK;
  int i0 = blockIdx.y * BLK;

  __shared__ uint16_t As[2][BLK * KT];
  __shared__ uint16_t Bg[2][BLK * KT];
  __shared__ uint16_t Bu[2][BLK * KT];
  __shared__ int stok[BLK];

  int tid = threadIdx.x;
  if (tid < BLK) stok[tid] = tok[e * T_TOK + tt * BLK + ((tid < nrows) ? tid : 0)];
  __syncthreads();

  int wid = tid >> 5, lane = tid & 31;
  int wr = wid & 3, wc = wid >> 2;
  int lrow = lane & 15, lsel = lane >> 4;

  int r = tid >> 1, half = tid & 1;
  size_t rowA = (size_t)stok[r] * H_DIM;
  size_t rowB = ((size_t)e * I_DIM + i0 + r) * H_DIM;
  int lofs = r * KT + half * 16;

  f32x8 accG[2][4] = {}, accU[2][4] = {};

  // prologue: stage k=0 tiles into buffer 0
  {
    int ke = half * 16;
    ld_async_b128(&As[0][lofs],     xbf  + rowA + ke);
    ld_async_b128(&As[0][lofs + 8], xbf  + rowA + ke + 8);
    ld_async_b128(&Bg[0][lofs],     wgbf + rowB + ke);
    ld_async_b128(&Bg[0][lofs + 8], wgbf + rowB + ke + 8);
    ld_async_b128(&Bu[0][lofs],     wubf + rowB + ke);
    ld_async_b128(&Bu[0][lofs + 8], wubf + rowB + ke + 8);
  }

  int p = 0;
  for (int k0 = 0; k0 < H_DIM; k0 += KT) {
    wait_async0();          // own async writes into buf p complete
    __syncthreads();        // all waves' buf-p tiles visible; buf p^1 readers done
    if (k0 + KT < H_DIM) {  // overlap: DMA next tile while computing this one
      int ke = k0 + KT + half * 16;
      ld_async_b128(&As[p ^ 1][lofs],     xbf  + rowA + ke);
      ld_async_b128(&As[p ^ 1][lofs + 8], xbf  + rowA + ke + 8);
      ld_async_b128(&Bg[p ^ 1][lofs],     wgbf + rowB + ke);
      ld_async_b128(&Bg[p ^ 1][lofs + 8], wgbf + rowB + ke + 8);
      ld_async_b128(&Bu[p ^ 1][lofs],     wubf + rowB + ke);
      ld_async_b128(&Bu[p ^ 1][lofs + 8], wubf + rowB + ke + 8);
    }

    FragAB a[2], bg[4], bu[4];
    for (int mt = 0; mt < 2; mt++) {
      const uint16_t* pa = &As[p][(wr * 32 + mt * 16 + lrow) * KT + lsel * 16];
      a[mt].u[0] = *(const u32x4*)pa;
      a[mt].u[1] = *(const u32x4*)(pa + 8);
    }
    for (int nt = 0; nt < 4; nt++) {
      const uint16_t* pg = &Bg[p][(wc * 64 + nt * 16 + lrow) * KT + lsel * 16];
      bg[nt].u[0] = *(const u32x4*)pg;
      bg[nt].u[1] = *(const u32x4*)(pg + 8);
      const uint16_t* pu = &Bu[p][(wc * 64 + nt * 16 + lrow) * KT + lsel * 16];
      bu[nt].u[0] = *(const u32x4*)pu;
      bu[nt].u[1] = *(const u32x4*)(pu + 8);
    }
    for (int mt = 0; mt < 2; mt++)
      for (int nt = 0; nt < 4; nt++) {
        accG[mt][nt] = __builtin_amdgcn_wmma_f32_16x16x32_bf16(
            false, a[mt].bf, false, bg[nt].bf, (short)0, accG[mt][nt], false, false);
        accU[mt][nt] = __builtin_amdgcn_wmma_f32_16x16x32_bf16(
            false, a[mt].bf, false, bu[nt].bf, (short)0, accU[mt][nt], false, false);
      }
    p ^= 1;
  }

  // epilogue: act = silu(g) * u, store bf16
  for (int mt = 0; mt < 2; mt++)
    for (int nt = 0; nt < 4; nt++)
      for (int j = 0; j < 8; j++) {
        int rl = wr * 32 + mt * 16 + j + 8 * lsel;   // C layout: VGPR j -> M = j + 8*(lane>>4)
        if (rl < nrows) {
          float g = accG[mt][nt][j], u = accU[mt][nt][j];
          float s = g / (1.f + __expf(-g));
          int col = wc * 64 + nt * 16 + lrow;
          act[((size_t)(e * T_TOK + tt * BLK + rl)) * I_DIM + i0 + col] =
              f32_to_bf16(s * u);
        }
      }
}

// ---------------- GEMM2: act @ wd^T, scale by routing weight, scatter-add ----
__global__ __launch_bounds__(256) void k_gemm2(
    const uint16_t* __restrict__ act, const uint16_t* __restrict__ wdbf,
    const int* __restrict__ cnt, const int* __restrict__ tok,
    const float* __restrict__ wt, float* __restrict__ out) {
  int e  = blockIdx.x % E_NUM;
  int tt = blockIdx.x / E_NUM;
  int nrows = cnt[e] - tt * BLK;
  if (nrows <= 0) return;
  if (nrows > BLK) nrows = BLK;
  int h0 = blockIdx.y * BLK;

  __shared__ uint16_t As[2][BLK * KT];
  __shared__ uint16_t Bd[2][BLK * KT];
  __shared__ int   stok[BLK];
  __shared__ float swt[BLK];

  int tid = threadIdx.x;
  if (tid < BLK) {
    bool ok = tid < nrows;
    stok[tid] = ok ? tok[e * T_TOK + tt * BLK + tid] : 0;
    swt[tid]  = ok ? wt[e * T_TOK + tt * BLK + tid] : 0.f;
  }
  __syncthreads();

  int wid = tid >> 5, lane = tid & 31;
  int wr = wid & 3, wc = wid >> 2;
  int lrow = lane & 15, lsel = lane >> 4;

  int r = tid >> 1, half = tid & 1;
  size_t rowA = ((size_t)(e * T_TOK + tt * BLK + ((r < nrows) ? r : 0))) * I_DIM;
  size_t rowB = ((size_t)e * H_DIM + h0 + r) * I_DIM;
  int lofs = r * KT + half * 16;

  f32x8 acc[2][4] = {};

  {
    int ke = half * 16;
    ld_async_b128(&As[0][lofs],     act  + rowA + ke);
    ld_async_b128(&As[0][lofs + 8], act  + rowA + ke + 8);
    ld_async_b128(&Bd[0][lofs],     wdbf + rowB + ke);
    ld_async_b128(&Bd[0][lofs + 8], wdbf + rowB + ke + 8);
  }

  int p = 0;
  for (int k0 = 0; k0 < I_DIM; k0 += KT) {
    wait_async0();
    __syncthreads();
    if (k0 + KT < I_DIM) {
      int ke = k0 + KT + half * 16;
      ld_async_b128(&As[p ^ 1][lofs],     act  + rowA + ke);
      ld_async_b128(&As[p ^ 1][lofs + 8], act  + rowA + ke + 8);
      ld_async_b128(&Bd[p ^ 1][lofs],     wdbf + rowB + ke);
      ld_async_b128(&Bd[p ^ 1][lofs + 8], wdbf + rowB + ke + 8);
    }

    FragAB a[2], b[4];
    for (int mt = 0; mt < 2; mt++) {
      const uint16_t* pa = &As[p][(wr * 32 + mt * 16 + lrow) * KT + lsel * 16];
      a[mt].u[0] = *(const u32x4*)pa;
      a[mt].u[1] = *(const u32x4*)(pa + 8);
    }
    for (int nt = 0; nt < 4; nt++) {
      const uint16_t* pb = &Bd[p][(wc * 64 + nt * 16 + lrow) * KT + lsel * 16];
      b[nt].u[0] = *(const u32x4*)pb;
      b[nt].u[1] = *(const u32x4*)(pb + 8);
    }
    for (int mt = 0; mt < 2; mt++)
      for (int nt = 0; nt < 4; nt++)
        acc[mt][nt] = __builtin_amdgcn_wmma_f32_16x16x32_bf16(
            false, a[mt].bf, false, b[nt].bf, (short)0, acc[mt][nt], false, false);
    p ^= 1;
  }

  for (int mt = 0; mt < 2; mt++)
    for (int nt = 0; nt < 4; nt++)
      for (int j = 0; j < 8; j++) {
        int rl = wr * 32 + mt * 16 + j + 8 * lsel;
        if (rl < nrows) {
          int col = wc * 64 + nt * 16 + lrow;
          float v = acc[mt][nt][j] * swt[rl];
          // no-return f32 atomic add (global_atomic_add_f32), relaxed, device scope
          (void)__hip_atomic_fetch_add(&out[(size_t)stok[rl] * H_DIM + h0 + col], v,
                                       __ATOMIC_RELAXED, __HIP_MEMORY_SCOPE_AGENT);
        }
      }
}

// ---------------- host-side launch ----------------
extern "C" void kernel_launch(void* const* d_in, const int* in_sizes, int n_in,
                              void* d_out, int out_size, void* d_ws, size_t ws_size,
                              hipStream_t stream) {
  (void)in_sizes; (void)n_in; (void)out_size; (void)ws_size;
  const float* x  = (const float*)d_in[0];
  const float* gw = (const float*)d_in[1];
  const float* wg = (const float*)d_in[2];
  const float* wu = (const float*)d_in[3];
  const float* wd = (const float*)d_in[4];
  float* out = (float*)d_out;

  char* w = (char*)d_ws;
  int*      cnt  = (int*)w;        w += 256;
  int*      tok  = (int*)w;        w += (size_t)E_NUM * T_TOK * 4;
  float*    wts  = (float*)w;      w += (size_t)E_NUM * T_TOK * 4;
  float*    comb = (float*)w;      w += (size_t)T_TOK * E_NUM * 4;
  uint16_t* xbf  = (uint16_t*)w;   w += (size_t)T_TOK * H_DIM * 2;
  uint16_t* wgbf = (uint16_t*)w;   w += (size_t)E_NUM * I_DIM * H_DIM * 2;
  uint16_t* wubf = (uint16_t*)w;   w += (size_t)E_NUM * I_DIM * H_DIM * 2;
  uint16_t* wdbf = (uint16_t*)w;   w += (size_t)E_NUM * H_DIM * I_DIM * 2;
  uint16_t* act  = (uint16_t*)w;   w += (size_t)E_NUM * T_TOK * I_DIM * 2;

  k_zero<<<4096, 256, 0, stream>>>(out, T_TOK * H_DIM / 4);
  k_cvt_bf16<<<2048, 256, 0, stream>>>(x,  xbf,  T_TOK * H_DIM / 4);
  k_cvt_bf16<<<4096, 256, 0, stream>>>(wg, wgbf, E_NUM * I_DIM * H_DIM / 4);
  k_cvt_bf16<<<4096, 256, 0, stream>>>(wu, wubf, E_NUM * I_DIM * H_DIM / 4);
  k_cvt_bf16<<<4096, 256, 0, stream>>>(wd, wdbf, E_NUM * H_DIM * I_DIM / 4);
  k_router<<<T_TOK, 256, 0, stream>>>(x, gw, out + (size_t)T_TOK * H_DIM, comb);
  k_compact<<<E_NUM, 32, 0, stream>>>(comb, cnt, tok, wts);
  k_gemm1<<<dim3(E_NUM * (T_TOK / BLK), I_DIM / BLK), 256, 0, stream>>>(
      xbf, wgbf, wubf, cnt, tok, act);
  k_gemm2<<<dim3(E_NUM * (T_TOK / BLK), H_DIM / BLK), 256, 0, stream>>>(
      act, wdbf, cnt, tok, wts, out);
}
